// GAT_42649025249475
// MI455X (gfx1250) — matrix operand order repaired and checked by reading, not compiled, not run on hardware
//
#include <hip/hip_runtime.h>

#define NN      8192
#define NEMB    128
#define NHID    64
#define NCLASS  16
#define NWORDS  (NN / 32)   // 256 mask words per row

typedef __attribute__((ext_vector_type(16))) _Float16 v16h;
typedef __attribute__((ext_vector_type(8)))  float    v8f;
typedef __attribute__((ext_vector_type(4)))  float    v4f;
typedef __attribute__((ext_vector_type(2)))  float    v2f;

#define LOG2E 1.4426950408889634f

// ---------------------------------------------------------------------------
// adj (dense f32 0/1) -> 1-bit mask words. One wave per row; __ballot packs 32
// columns per iteration. Nontemporal loads + prefetch: adj is touched exactly
// once from HBM for the whole 2-layer network (268 MB -> 8 MB, L2-resident).
// ---------------------------------------------------------------------------
__global__ void k_mask(const float* __restrict__ adj, unsigned* __restrict__ mb) {
  const int lane = threadIdx.x & 31;
  const int row  = (blockIdx.x * blockDim.x + threadIdx.x) >> 5;
  const float* rp = adj + (size_t)row * NN;
  unsigned* out = mb + (size_t)row * NWORDS;
  for (int w = 0; w < NWORDS; ++w) {
    float v = __builtin_nontemporal_load(rp + w * 32 + lane);
    int pfw = (w + 16 < NWORDS) ? w + 16 : w;
    __builtin_prefetch(rp + pfw * 32 + lane, 0, 0);   // global_prefetch_b8
    unsigned m = (unsigned)__ballot(v > 0.f);         // wave32 ballot
    if (lane == 0) out[w] = m;
  }
}

// ---------------------------------------------------------------------------
// out[NN,F] = A[NN,K] @ W[K,F] on the native f32 matrix pipe:
// V_WMMA_F32_16X16X4_F32, one wave per 16x16 output tile, K/4 matrix ops.
// A(16x4) layout: lane = m + 16*(k>=2), vgpr = k%2 -> one float2 load/lane.
// B(4x16) layout: lane = n + 16*(k>=2), vgpr = k%2 -> two stride-F loads.
// C/D layout: row = vgpr + 8*(lane>=16), col = lane%16.
// ---------------------------------------------------------------------------
template<int K, int F>
__global__ void __launch_bounds__(32) k_gemm_wmma(const float* __restrict__ A,
                                                  const float* __restrict__ W,
                                                  float* __restrict__ out) {
  const int lane = threadIdx.x;
  const int g    = lane >> 4;
  const int mrow = lane & 15;
  const int rb   = blockIdx.x * 16;
  const int cb   = blockIdx.y * 16;
  const float* arow = A + (size_t)(rb + mrow) * K + 2 * g;
  const float* wcol = W + (size_t)(2 * g) * F + cb + mrow;
  v8f acc = {};
#pragma unroll
  for (int kb = 0; kb < K; kb += 4) {
    v2f av = *(const v2f*)(arow + kb);
    v2f bv;
    bv.x = wcol[(size_t)kb * F];
    bv.y = wcol[(size_t)kb * F + F];
    acc = __builtin_amdgcn_wmma_f32_16x16x4_f32(false, av, false, bv,
                                                (short)0, acc, false, false);
  }
#pragma unroll
  for (int v = 0; v < 8; ++v)
    out[(size_t)(rb + v + 8 * g) * F + cb + mrow] = acc[v];
}

// ---------------------------------------------------------------------------
// Per-node: s = h.a_src, d = h.a_dst, global dmax (ordered-uint atomicMax),
// and pack H into the exact CDNA5 16x16x32 f16 B-operand lane layout:
//   element (k,n): lane = n + 16*((k/8)%2), per-lane f16 index = k%8 + 8*(k>=16),
//   region per (jt, nt) = 512 f16 = 32 lanes * 16.
// ---------------------------------------------------------------------------
template<int F>
__global__ void k_stats(const float* __restrict__ H, const float* __restrict__ asrc,
                        const float* __restrict__ adst, float* __restrict__ s,
                        float* __restrict__ d, unsigned* __restrict__ dmax_enc,
                        _Float16* __restrict__ Bp) {
  constexpr int NT = F / 16;
  int j = blockIdx.x * blockDim.x + threadIdx.x;
  const float* h = H + (size_t)j * F;
  int jt = j >> 5, k = j & 31;
  int laneHalf = ((k >> 3) & 1) << 4;   // +16 lanes for k%16 >= 8
  int kk = (k & 7) | ((k >> 4) << 3);   // per-lane element index
  float sv = 0.f, dv = 0.f;
#pragma unroll
  for (int f = 0; f < F; ++f) {
    float hv = h[f];
    sv = fmaf(hv, asrc[f], sv);
    dv = fmaf(hv, adst[f], dv);
    int nt = f >> 4, n = f & 15;
    Bp[((size_t)(jt * NT + nt)) * 512 + (size_t)(n + laneHalf) * 16 + kk] = (_Float16)hv;
  }
  s[j] = sv;
  d[j] = dv;
  unsigned u = __float_as_uint(dv);              // order-preserving f32->u32
  u = (u & 0x80000000u) ? ~u : (u | 0x80000000u);
  atomicMax(dmax_enc, u);
}

// ---------------------------------------------------------------------------
// Fused masked-softmax attention aggregation:
//   num[i,:] += sum_j exp(lrelu(s_i+d_j) - m_i) * mask_ij * H[j,:]   (f16 WMMA)
//   l[i]     += sum_j p_ij
// m_i = lrelu(s_i + dmax) is a per-row logit upper bound (lrelu monotonic), so
// no online rescaling is needed; shift-invariance keeps softmax exact.
// Per element (branchless): add, mul+max (lrelu), fma+exp2 (shift folded into
// exponent), int-AND mask zeroing, f16 cvt, lsum add.
// ---------------------------------------------------------------------------
template<int F, int RT, int JSPLIT>
__global__ void __launch_bounds__(32) k_attn(
    const unsigned* __restrict__ mb, const _Float16* __restrict__ Bp,
    const float* __restrict__ s, const float* __restrict__ d,
    const unsigned* __restrict__ dmax_enc,
    float* __restrict__ num, float* __restrict__ lden) {
  constexpr int NT = F / 16;
  constexpr int JT_PER = NWORDS / JSPLIT;
  const int lane = threadIdx.x;
  const int g    = lane >> 4;       // half-wave group
  const int mrow = lane & 15;       // A-matrix row owned by this lane
  const int rb   = blockIdx.x * (16 * RT);
  const int kq   = g * 8;

  unsigned u = *dmax_enc;
  float dmax = (u & 0x80000000u) ? __uint_as_float(u & 0x7fffffffu)
                                 : __uint_as_float(~u);

  float si[RT], nml[RT], ls[RT];
  const unsigned* mrp[RT];
#pragma unroll
  for (int rt = 0; rt < RT; ++rt) {
    int r = rb + rt * 16 + mrow;
    si[rt] = s[r];
    float t = si[rt] + dmax;
    float m = fmaxf(t, 0.2f * t);       // lrelu upper bound on row logits
    nml[rt] = -m * LOG2E;               // folded softmax shift (log2 domain)
    ls[rt] = 0.f;
    mrp[rt] = mb + (size_t)r * NWORDS;
  }
  v8f acc[RT][NT] = {};
  const int jt0 = blockIdx.y * JT_PER;

  for (int jt = jt0; jt < jt0 + JT_PER; ++jt) {
    const float* dp = d + (jt << 5) + kq;
    v4f d0 = *(const v4f*)(dp);
    v4f d1 = *(const v4f*)(dp + 4);
    v4f d2 = *(const v4f*)(dp + 16);
    v4f d3 = *(const v4f*)(dp + 20);

    v16h Am[RT];
#pragma unroll
    for (int rt = 0; rt < RT; ++rt) {
      const unsigned mw = mrp[rt][jt];
      const float sv = si[rt], nm = nml[rt];
      float lacc = 0.f;
#pragma unroll
      for (int e = 0; e < 4; ++e) {
        float t0 = sv + d0[e], t1 = sv + d1[e], t2 = sv + d2[e], t3 = sv + d3[e];
        float e0 = fmaxf(t0, 0.2f * t0);
        float e1 = fmaxf(t1, 0.2f * t1);
        float e2 = fmaxf(t2, 0.2f * t2);
        float e3 = fmaxf(t3, 0.2f * t3);
        float x0 = __builtin_amdgcn_exp2f(fmaf(e0, LOG2E, nm));
        float x1 = __builtin_amdgcn_exp2f(fmaf(e1, LOG2E, nm));
        float x2 = __builtin_amdgcn_exp2f(fmaf(e2, LOG2E, nm));
        float x3 = __builtin_amdgcn_exp2f(fmaf(e3, LOG2E, nm));
        // branchless zeroing: keep-mask = 0xFFFFFFFF iff adjacency bit set
        unsigned k0 = 0u - ((mw >> (kq + e))          & 1u);
        unsigned k1 = 0u - ((mw >> (kq + 4 + e))      & 1u);
        unsigned k2 = 0u - ((mw >> (16 + kq + e))     & 1u);
        unsigned k3 = 0u - ((mw >> (16 + kq + 4 + e)) & 1u);
        float p0 = __uint_as_float(__float_as_uint(x0) & k0);
        float p1 = __uint_as_float(__float_as_uint(x1) & k1);
        float p2 = __uint_as_float(__float_as_uint(x2) & k2);
        float p3 = __uint_as_float(__float_as_uint(x3) & k3);
        lacc += (p0 + p1) + (p2 + p3);
        Am[rt][e]      = (_Float16)p0;
        Am[rt][4 + e]  = (_Float16)p1;
        Am[rt][8 + e]  = (_Float16)p2;
        Am[rt][12 + e] = (_Float16)p3;
      }
      ls[rt] += lacc;
    }

    const _Float16* Bb = Bp + (size_t)jt * (NT * 512) + (size_t)lane * 16;
#pragma unroll
    for (int nt = 0; nt < NT; ++nt) {
      v16h Bm = *(const v16h*)(Bb + (size_t)nt * 512);
#pragma unroll
      for (int rt = 0; rt < RT; ++rt)
        acc[rt][nt] = __builtin_amdgcn_wmma_f32_16x16x32_f16(
            false, Am[rt], false, Bm, (short)0, acc[rt][nt], false, false);
    }
  }

#pragma unroll
  for (int rt = 0; rt < RT; ++rt) {
    float t = ls[rt] + __shfl_xor(ls[rt], 16, 32);   // combine K-halves per row
    if (lane < 16) atomicAdd(&lden[rb + rt * 16 + lane], t);
  }
  // C/D layout: row = v + 8g, col = lane%16
#pragma unroll
  for (int rt = 0; rt < RT; ++rt)
#pragma unroll
    for (int nt = 0; nt < NT; ++nt)
#pragma unroll
      for (int v = 0; v < 8; ++v)
        atomicAdd(&num[(size_t)(rb + rt * 16 + v + g * 8) * F + nt * 16 + mrow],
                  acc[rt][nt][v]);
}

// ---------------------------------------------------------------------------
// out = ELU(num / l + bias)
// ---------------------------------------------------------------------------
template<int F>
__global__ void k_finalize(const float* __restrict__ num, const float* __restrict__ l,
                           const float* __restrict__ bias, float* __restrict__ out) {
  int idx = blockIdx.x * blockDim.x + threadIdx.x;
  int i = idx / F, f = idx % F;
  float v = num[idx] / l[i] + bias[f];
  out[idx] = v > 0.f ? v : (__expf(v) - 1.f);
}

// ---------------------------------------------------------------------------
extern "C" void kernel_launch(void* const* d_in, const int* in_sizes, int n_in,
                              void* d_out, int out_size, void* d_ws, size_t ws_size,
                              hipStream_t stream) {
  (void)in_sizes; (void)n_in; (void)out_size; (void)ws_size;
  const float* x   = (const float*)d_in[0];
  const float* adj = (const float*)d_in[1];
  const float* W1  = (const float*)d_in[2];
  const float* a1s = (const float*)d_in[3];
  const float* a1d = (const float*)d_in[4];
  const float* b1  = (const float*)d_in[5];
  const float* W2  = (const float*)d_in[6];
  const float* a2s = (const float*)d_in[7];
  const float* a2d = (const float*)d_in[8];
  const float* b2  = (const float*)d_in[9];

  char* ws = (char*)d_ws;
  size_t off = 0;
  auto alloc = [&](size_t bytes) -> char* {
    char* p = ws + off;
    off = (off + bytes + 255) & ~(size_t)255;
    return p;
  };
  float*    H    = (float*)alloc((size_t)NN * NHID * 4);      // H1 / H2
  _Float16* Bp   = (_Float16*)alloc((size_t)NN * NHID * 2);   // packed B operand
  float*    svec = (float*)alloc((size_t)NN * 4);
  float*    dvec = (float*)alloc((size_t)NN * 4);
  float*    num  = (float*)alloc((size_t)NN * NHID * 4);
  float*    lvec = (float*)alloc((size_t)NN * 4);
  float*    Hact = (float*)alloc((size_t)NN * NHID * 4);      // layer-1 output
  unsigned* dmax = (unsigned*)alloc(256);
  unsigned* mb   = (unsigned*)alloc((size_t)NN * NWORDS * 4); // 8 MB bitmask

  constexpr int RT = 2, JSPLIT = 16;

  // adj -> bitmask (sole HBM stream of the 268 MB adjacency)
  k_mask<<<NN / 8, 256, 0, stream>>>(adj, mb);

  // ---- Layer 1 ----
  k_gemm_wmma<NEMB, NHID><<<dim3(NN / 16, NHID / 16), 32, 0, stream>>>(x, W1, H);
  hipMemsetAsync(num, 0, (size_t)NN * NHID * 4, stream);
  hipMemsetAsync(lvec, 0, (size_t)NN * 4, stream);
  hipMemsetAsync(dmax, 0, 4, stream);
  k_stats<NHID><<<NN / 256, 256, 0, stream>>>(H, a1s, a1d, svec, dvec, dmax, Bp);
  k_attn<NHID, RT, JSPLIT><<<dim3(NN / (16 * RT), JSPLIT), 32, 0, stream>>>(
      mb, Bp, svec, dvec, dmax, num, lvec);
  k_finalize<NHID><<<(size_t)NN * NHID / 256, 256, 0, stream>>>(num, lvec, b1, Hact);

  // ---- Layer 2 ----
  k_gemm_wmma<NHID, NCLASS><<<dim3(NN / 16, NCLASS / 16), 32, 0, stream>>>(Hact, W2, H);
  hipMemsetAsync(num, 0, (size_t)NN * NCLASS * 4, stream);
  hipMemsetAsync(lvec, 0, (size_t)NN * 4, stream);
  hipMemsetAsync(dmax, 0, 4, stream);
  k_stats<NCLASS><<<NN / 256, 256, 0, stream>>>(H, a2s, a2d, svec, dvec, dmax, Bp);
  k_attn<NCLASS, RT, JSPLIT><<<dim3(NN / (16 * RT), JSPLIT), 32, 0, stream>>>(
      mb, Bp, svec, dvec, dmax, num, lvec);
  k_finalize<NCLASS><<<(size_t)NN * NCLASS / 256, 256, 0, stream>>>(
      num, lvec, b2, (float*)d_out);
}
